// DynamicReductionNetwork_3453153706713
// MI455X (gfx1250) — matrix operand order repaired and checked by reading, not compiled, not run on hardware
//
#include <hip/hip_runtime.h>
#include <hip/hip_bf16.h>

// ---------------- CDNA5 WMMA types ----------------
typedef __attribute__((ext_vector_type(16))) _Float16 v16h;
typedef __attribute__((ext_vector_type(8)))  _Float16 v8h;
typedef __attribute__((ext_vector_type(8)))  float    v8f;
typedef _Float16 half_t;

#define NEG_SLOPE 0.4f
#define N0      8192
#define NG      16
#define K_NN    16
#define ADJ_W   256          // 8192 bits / 32 per adjacency row
#define FPITCH  256          // float pitch for all feature buffers
#define MAXE    262144       // <= 2 * K_NN * N0 undirected edges

__device__ __forceinline__ float lrelu(float v) { return v > 0.f ? v : NEG_SLOPE * v; }

// convert 8 contiguous fp32 -> a[BASE..BASE+7] (two aligned float4 loads)
template <int BASE>
__device__ __forceinline__ void cvt8(const float* __restrict__ p, v16h& a) {
    float4 u = ((const float4*)p)[0];
    float4 w = ((const float4*)p)[1];
    a[BASE + 0] = (half_t)u.x; a[BASE + 1] = (half_t)u.y;
    a[BASE + 2] = (half_t)u.z; a[BASE + 3] = (half_t)u.w;
    a[BASE + 4] = (half_t)w.x; a[BASE + 5] = (half_t)w.y;
    a[BASE + 6] = (half_t)w.z; a[BASE + 7] = (half_t)w.w;
}
// convert 8 contiguous (pj - pi) differences -> a[BASE..]
template <int BASE>
__device__ __forceinline__ void cvt8d(const float* __restrict__ pj,
                                      const float* __restrict__ pi, v16h& a) {
    float4 j0 = ((const float4*)pj)[0], j1 = ((const float4*)pj)[1];
    float4 i0 = ((const float4*)pi)[0], i1 = ((const float4*)pi)[1];
    a[BASE + 0] = (half_t)(j0.x - i0.x); a[BASE + 1] = (half_t)(j0.y - i0.y);
    a[BASE + 2] = (half_t)(j0.z - i0.z); a[BASE + 3] = (half_t)(j0.w - i0.w);
    a[BASE + 4] = (half_t)(j1.x - i1.x); a[BASE + 5] = (half_t)(j1.y - i1.y);
    a[BASE + 6] = (half_t)(j1.z - i1.z); a[BASE + 7] = (half_t)(j1.w - i1.w);
}
__device__ __forceinline__ void zero16(v16h& a) {
#pragma unroll
    for (int t = 0; t < 16; t++) a[t] = (half_t)0.f;
}

// ---------------- workspace layout (bytes) ----------------
constexpr size_t OFF_META   = 0;
constexpr size_t OFF_GSA    = 256;
constexpr size_t OFF_GSB    = 512;
constexpr size_t OFF_BATCHA = 768;
constexpr size_t OFF_BATCHB = OFF_BATCHA + (size_t)N0 * 4;
constexpr size_t OFF_DEG    = OFF_BATCHB + (size_t)N0 * 4;
constexpr size_t OFF_ROWPTR = OFF_DEG    + (size_t)N0 * 4;
constexpr size_t OFF_MATCH  = OFF_ROWPTR + (size_t)8448 * 4;
constexpr size_t OFF_CID    = OFF_MATCH  + (size_t)N0 * 4;
constexpr size_t OFF_MEM1   = OFF_CID    + (size_t)N0 * 4;
constexpr size_t OFF_MEM2   = OFF_MEM1   + (size_t)N0 * 4;
constexpr size_t OFF_GBUF   = OFF_MEM2   + (size_t)N0 * 4;
constexpr size_t OFF_XN     = OFF_GBUF   + (size_t)NG * 64 * 4;      // pitch-32 padded input
constexpr size_t OFF_EROW   = OFF_XN     + (size_t)N0 * 32 * 4;
constexpr size_t OFF_ECOL   = OFF_EROW   + (size_t)MAXE * 4;
constexpr size_t OFF_EW     = OFF_ECOL   + (size_t)MAXE * 4;
constexpr size_t OFF_WP     = OFF_EW     + (size_t)MAXE * 4;         // packed f16 weights
constexpr size_t OFF_ADJ    = OFF_WP     + (size_t)524288;
constexpr size_t OFF_F0     = OFF_ADJ    + (size_t)N0 * ADJ_W * 4;
constexpr size_t OFF_F1     = OFF_F0     + (size_t)N0 * FPITCH * 4;
constexpr size_t OFF_F2     = OFF_F1     + (size_t)N0 * FPITCH * 4;
constexpr size_t WS_NEED    = OFF_F2     + (size_t)N0 * FPITCH * 4;  // ~38 MB

// ---------------- small helpers ----------------
__global__ void k_init(int* meta, int* gsA) {
    int t = threadIdx.x;
    if (t == 0) { meta[0] = N0; meta[1] = 0; }
    if (t <= NG) gsA[t] = t * (N0 / NG);
}

__global__ void k_copyint(int* dst, const int* src, int n) {
    int i = blockIdx.x * blockDim.x + threadIdx.x;
    if (i < n) dst[i] = src[i];
}

__global__ void k_scale(float* xn, const float* x, const float* dn, int n) {
    int i = blockIdx.x * blockDim.x + threadIdx.x;
    if (i < n) xn[(i / 5) * 32 + (i % 5)] = x[i] * dn[i % 5];
}

// pack fp32 W[KxN] -> f16 WMMA B-operand layout: [kchunk][ntile][lane 0..31][16 halves]
__global__ void k_pack(const float* __restrict__ W, int K, int Kpad, int N,
                       half_t* __restrict__ P) {
    int idx = blockIdx.x * blockDim.x + threadIdx.x;
    if (idx >= Kpad * N) return;
    int t      = idx & 15;
    int lane   = (idx >> 4) & 31;
    int tile   = idx >> 9;
    int ntiles = N >> 4;
    int tn     = tile % ntiles;
    int kchunk = tile / ntiles;
    int k   = kchunk * 32 + ((t >> 3) << 4) + ((lane >> 4) << 3) + (t & 7);
    int col = tn * 16 + (lane & 15);
    P[idx] = (k < K) ? (half_t)W[(size_t)k * N + col] : (half_t)0.f;
}

// ---------------- generic WMMA GEMM: C = act(A[MxK] @ W[KxN] + b) ----------------
// K must be a multiple of 32 (callers pad). Out-of-range rows are clamped (not stored).
__global__ void k_gemm(const float* __restrict__ A, int lda,
                       const half_t* __restrict__ Bp, const float* __restrict__ bias,
                       float* __restrict__ C, int ldc,
                       const int* __restrict__ meta, int K, int N, int relu) {
    int M = meta[0];
    int tm = blockIdx.x, tn = blockIdx.y;
    if (tm * 16 >= M) return;
    int lane = threadIdx.x;
    int mr = lane & 15, hl = lane >> 4;
    int row = tm * 16 + mr; if (row > M - 1) row = M - 1;
    int col = tn * 16 + mr;
    const int ntiles = N >> 4;
    const float* Ar = A + (size_t)row * lda;

    v8f acc = {};
    for (int kc = 0; kc < K; kc += 32) {
        v16h a;
        const float* p = Ar + kc + hl * 8;
        cvt8<0>(p, a);
        cvt8<8>(p + 16, a);
        v16h b = *(const v16h*)(Bp + (((size_t)(kc >> 5) * ntiles + tn) << 9) + (lane << 4));
        acc = __builtin_amdgcn_wmma_f32_16x16x32_f16(false, a, false, b,
                                                     (short)0, acc, false, false);
    }
    float bb = bias[col];
#pragma unroll
    for (int q = 0; q < 8; q++) {
        int r = tm * 16 + q + hl * 8;
        if (r < M) {
            float v = acc[q] + bb;
            if (relu) v = lrelu(v);
            C[(size_t)r * ldc + col] = v;
        }
    }
}

// ---------------- batch-aware kNN -> symmetric adjacency bitmask ----------------
__global__ void k_knn(const float* __restrict__ H, int ld, int F,
                      const int* __restrict__ batch, const int* __restrict__ gstart,
                      const int* __restrict__ meta, unsigned* __restrict__ adj) {
    int n = meta[0];
    int i = blockIdx.x;
    if (i >= n) return;
    int lane = threadIdx.x;
    int g = batch[i];
    int gs = gstart[g], ge = gstart[g + 1];
    int cnt = ge - gs; if (cnt > 512) cnt = 512;
    const float FINF = __builtin_inff();
    const float* Hi = H + (size_t)i * ld;
    __shared__ float dist[512];

    for (int jj = lane; jj < cnt; jj += 32) {
        int j = gs + jj;
        float d;
        if (j == i) d = FINF;
        else {
            const float* Hj = H + (size_t)j * ld;
            d = 0.f;
            for (int f = 0; f < F; f += 4) {
                float4 aa = *(const float4*)(Hi + f);
                float4 bb = *(const float4*)(Hj + f);
                float t0 = aa.x - bb.x, t1 = aa.y - bb.y, t2 = aa.z - bb.z, t3 = aa.w - bb.w;
                d += t0 * t0 + t1 * t1 + t2 * t2 + t3 * t3;
            }
        }
        dist[jj] = d;
    }
    __syncthreads();

    for (int t = 0; t < K_NN; t++) {
        float bv = FINF; int bi = -1;
        for (int jj = lane; jj < cnt; jj += 32) {
            float v = dist[jj];
            if (v < bv || (v == bv && (unsigned)jj < (unsigned)bi)) { bv = v; bi = jj; }
        }
        for (int off = 16; off > 0; off >>= 1) {       // wave32 min-reduce, low-index ties
            float ov = __shfl_xor(bv, off);
            int   oi = __shfl_xor(bi, off);
            if (ov < bv || (ov == bv && (unsigned)oi < (unsigned)bi)) { bv = ov; bi = oi; }
        }
        if (bi < 0 || !(bv < FINF)) break;             // uniform across wave
        if (lane == 0) {
            int j = gs + bi;
            atomicOr(&adj[(size_t)i * ADJ_W + (j >> 5)], 1u << (j & 31));
            atomicOr(&adj[(size_t)j * ADJ_W + (i >> 5)], 1u << (i & 31));
            dist[bi] = FINF;
        }
        __syncthreads();
    }
}

__global__ void k_deg(const unsigned* __restrict__ adj, int* __restrict__ deg,
                      const int* __restrict__ meta) {
    int n = meta[0];
    int i = blockIdx.x * blockDim.x + threadIdx.x;
    if (i >= n) return;
    int s = 0;
    for (int w = 0; w < ADJ_W; w++) s += __popc(adj[(size_t)i * ADJ_W + w]);
    deg[i] = s;
}

__global__ void k_scan(const int* __restrict__ deg, int* __restrict__ rowptr, int* meta) {
    if (threadIdx.x | blockIdx.x) return;
    int n = meta[0];
    rowptr[0] = 0;
    for (int i = 0; i < n; i++) rowptr[i + 1] = rowptr[i] + deg[i];
    meta[1] = rowptr[n];
}

__global__ void k_csr(const unsigned* __restrict__ adj, const int* __restrict__ rowptr,
                      int* __restrict__ erow, int* __restrict__ ecol,
                      const int* __restrict__ meta) {
    int n = meta[0];
    int i = blockIdx.x * blockDim.x + threadIdx.x;
    if (i >= n) return;
    int p = rowptr[i];
    for (int w = 0; w < ADJ_W; w++) {
        unsigned bits = adj[(size_t)i * ADJ_W + w];
        while (bits) {
            int b = __ffs(bits) - 1;
            erow[p] = i; ecol[p] = w * 32 + b; p++;
            bits &= bits - 1;
        }
    }
}

// ---------------- fused EdgeConv: sum_j lrelu(lrelu([x_i, x_j-x_i]W1+b1)W2+b2) ----------------
// one wave per 16 destination nodes; F is a multiple of 32 -> chunk-uniform operand source.
__global__ void k_edgeconv(const float* __restrict__ H, int ld, int F,
                           const half_t* __restrict__ W1p, const float* __restrict__ B1,
                           const half_t* __restrict__ W2p, const float* __restrict__ B2,
                           float* __restrict__ OUT, int ldo,
                           const int* __restrict__ rowptr, const int* __restrict__ ecol,
                           const int* __restrict__ meta) {
    int n = meta[0];
    int tile = blockIdx.x;
    if (tile * 16 >= n) return;
    int lane = threadIdx.x;
    int r = lane & 15, hl = lane >> 4;

    __shared__ half_t hid[16][96];
    __shared__ int sdeg[16], sbase[16], sj[16];

    if (lane < 16) {
        int node = tile * 16 + lane;
        if (node < n) { int b0 = rowptr[node]; sbase[lane] = b0; sdeg[lane] = rowptr[node + 1] - b0; }
        else          { sbase[lane] = 0; sdeg[lane] = 0; }
    }
    __syncthreads();
    int maxdeg = 0;
#pragma unroll
    for (int t = 0; t < 16; t++) maxdeg = sdeg[t] > maxdeg ? sdeg[t] : maxdeg;

    v8f accO[4] = {};
    const int K2 = 2 * F;
    const int i0 = tile * 16 + r;
    const float* Hi = H + (size_t)i0 * ld;

    for (int p = 0; p < maxdeg; p++) {
        if (lane < 16) sj[lane] = (p < sdeg[lane]) ? ecol[sbase[lane] + p] : -1;
        __syncthreads();
        int j0 = sj[r];
        const float* Hj = (j0 >= 0) ? H + (size_t)j0 * ld : Hi;

        // layer 1: 16 x K2 -> 16 x 96 (f16 into LDS)
        for (int ct = 0; ct < 6; ct++) {
            v8f acc = {};
            for (int kc = 0; kc < K2; kc += 32) {
                v16h a;
                if (j0 < 0) zero16(a);
                else if (kc + 32 <= F) {               // first half: x_i (chunk-uniform)
                    const float* pp = Hi + kc + hl * 8;
                    cvt8<0>(pp, a); cvt8<8>(pp + 16, a);
                } else {                               // second half: x_j - x_i
                    const float* pj = Hj + (kc - F) + hl * 8;
                    const float* pi = Hi + (kc - F) + hl * 8;
                    cvt8d<0>(pj, pi, a); cvt8d<8>(pj + 16, pi + 16, a);
                }
                v16h b = *(const v16h*)(W1p + (((size_t)(kc >> 5) * 6 + ct) << 9) + (lane << 4));
                acc = __builtin_amdgcn_wmma_f32_16x16x32_f16(false, a, false, b,
                                                             (short)0, acc, false, false);
            }
            int col = ct * 16 + r;
            float bb = B1[col];
#pragma unroll
            for (int q = 0; q < 8; q++)
                hid[q + hl * 8][col] = (half_t)lrelu(acc[q] + bb);
        }
        __syncthreads();

        // layer 2: 16 x 96 -> 16 x 64, masked accumulate
        for (int ct = 0; ct < 4; ct++) {
            v8f acc = {};
            for (int kc = 0; kc < 96; kc += 32) {
                v16h a;
                v8h lo = *(const v8h*)(&hid[r][kc + hl * 8]);
                v8h hi = *(const v8h*)(&hid[r][kc + 16 + hl * 8]);
#pragma unroll
                for (int t = 0; t < 8; t++) { a[t] = lo[t]; a[8 + t] = hi[t]; }
                v16h b = *(const v16h*)(W2p + (((size_t)(kc >> 5) * 4 + ct) << 9) + (lane << 4));
                acc = __builtin_amdgcn_wmma_f32_16x16x32_f16(false, a, false, b,
                                                             (short)0, acc, false, false);
            }
            int col = ct * 16 + r;
            float bb = B2[col];
#pragma unroll
            for (int q = 0; q < 8; q++) {
                int rr = q + hl * 8;
                if (p < sdeg[rr]) accO[ct][q] += lrelu(acc[q] + bb);
            }
        }
        __syncthreads();
    }

    for (int ct = 0; ct < 4; ct++) {
        int col = ct * 16 + r;
#pragma unroll
        for (int q = 0; q < 8; q++) {
            int node = tile * 16 + q + hl * 8;
            if (node < n) OUT[(size_t)node * ldo + col] = accO[ct][q];
        }
    }
}

__global__ void k_rescopy(float* __restrict__ OUT, int ldo,
                          const float* __restrict__ IN, int ldi, int F,
                          const int* __restrict__ meta) {
    int n = meta[0];
    int idx = blockIdx.x * blockDim.x + threadIdx.x;
    int i = idx / F, f = idx % F;
    if (i >= n) return;
    OUT[(size_t)i * ldo + 64 + f] = IN[(size_t)i * ldi + f];
}

// ---------------- normalized-cut edge weights ----------------
__global__ void k_weight(const float* __restrict__ H, int ld, int D,
                         const int* __restrict__ erow, const int* __restrict__ ecol,
                         const int* __restrict__ deg, float* __restrict__ ew,
                         const int* __restrict__ meta) {
    int ne = meta[1];
    int e = blockIdx.x * blockDim.x + threadIdx.x;
    if (e >= ne) return;
    int r = erow[e], c = ecol[e];
    const float* Hr = H + (size_t)r * ld;
    const float* Hc = H + (size_t)c * ld;
    float d = 0.f;
    for (int f = 0; f < D; f += 4) {
        float4 aa = *(const float4*)(Hr + f);
        float4 bb = *(const float4*)(Hc + f);
        float t0 = aa.x - bb.x, t1 = aa.y - bb.y, t2 = aa.z - bb.z, t3 = aa.w - bb.w;
        d += t0 * t0 + t1 * t1 + t2 * t2 + t3 * t3;
    }
    ew[e] = sqrtf(d) * (1.f / (float)deg[r] + 1.f / (float)deg[c]);
}

// ---------------- sequential greedy graclus + relabel (matches numpy semantics) ----------------
__global__ void k_graclus(const int* __restrict__ rowptr, const int* __restrict__ ecol,
                          const float* __restrict__ ew, int* meta,
                          int* match, int* cid, int* mem1, int* mem2,
                          const int* __restrict__ bCur, int* bNew, int* gsNew) {
    if (threadIdx.x | blockIdx.x) return;
    int n = meta[0];
    for (int i = 0; i < n; i++) match[i] = -1;
    for (int i = 0; i < n; i++) {
        if (match[i] >= 0) continue;
        int best = -1; float bw = -__builtin_inff();
        for (int e = rowptr[i]; e < rowptr[i + 1]; e++) {
            int j = ecol[e];
            if (j != i && match[j] < 0 && ew[e] > bw) { best = j; bw = ew[e]; }
        }
        match[i] = i;
        if (best >= 0) match[best] = i;
    }
    int nn = 0;
    for (int i = 0; i < n; i++) {
        if (match[i] == i) { cid[i] = nn; mem1[nn] = i; mem2[nn] = -1; bNew[nn] = bCur[i]; nn++; }
    }
    for (int i = 0; i < n; i++)
        if (match[i] != i) mem2[cid[match[i]]] = i;
    int idx = 0;
    for (int gg = 0; gg < NG; gg++) {
        gsNew[gg] = idx;
        while (idx < nn && bNew[idx] == gg) idx++;
    }
    gsNew[NG] = nn;
    meta[0] = nn;
}

__global__ void k_pool(const float* __restrict__ IN, int ldi, int D,
                       const int* __restrict__ mem1, const int* __restrict__ mem2,
                       float* __restrict__ OUT, int ldo, const int* __restrict__ meta) {
    int n = meta[0];
    int idx = blockIdx.x * blockDim.x + threadIdx.x;
    int c = idx / D, f = idx % D;
    if (c >= n) return;
    int m1 = mem1[c], m2 = mem2[c];
    float v = IN[(size_t)m1 * ldi + f];
    if (m2 >= 0) v = fmaxf(v, IN[(size_t)m2 * ldi + f]);
    OUT[(size_t)c * ldo + f] = v;
}

__global__ void k_graphmax(const float* __restrict__ H, int ld,
                           const int* __restrict__ gstart, float* __restrict__ G) {
    int g = blockIdx.x, f = threadIdx.x;
    int gs = gstart[g], ge = gstart[g + 1];
    float m = -__builtin_inff();
    for (int i = gs; i < ge; i++) m = fmaxf(m, H[(size_t)i * ld + f]);
    G[g * 64 + f] = m;
}

__global__ void k_outmlp(const float* __restrict__ G,
                         const float* W1, const float* B1,
                         const float* W2, const float* B2,
                         const float* W3, const float* B3,
                         float* __restrict__ out) {
    int g = blockIdx.x, t = threadIdx.x;
    __shared__ float gin[64], h1[64], h2[32];
    gin[t] = G[g * 64 + t];
    __syncthreads();
    float s = B1[t];
    for (int k = 0; k < 64; k++) s += gin[k] * W1[k * 64 + t];
    h1[t] = lrelu(s);
    __syncthreads();
    if (t < 32) {
        float s2 = B2[t];
        for (int k = 0; k < 64; k++) s2 += h1[k] * W2[k * 32 + t];
        h2[t] = lrelu(s2);
    }
    __syncthreads();
    if (t == 0) {
        float s3 = B3[0];
        for (int k = 0; k < 32; k++) s3 += h2[k] * W3[k];
        out[g] = s3;
    }
}

// ---------------- host orchestration ----------------
extern "C" void kernel_launch(void* const* d_in, const int* in_sizes, int n_in,
                              void* d_out, int out_size, void* d_ws, size_t ws_size,
                              hipStream_t stream) {
    if (ws_size < WS_NEED) return;

    const float* x        = (const float*)d_in[0];
    const int*   batch_in = (const int*)d_in[1];

    // params pytree ordering: detect sorted-key flatten vs insertion-order flatten
    int iC1, iC2, iC3, iC4, iDN, iIN, iOUT;
    if (n_in >= 31 && in_sizes[2] == 128 * 96) {        // sorted: conv1..4, datanorm, inputnet, output
        iC1 = 2; iC2 = 6; iC3 = 10; iC4 = 14; iDN = 18; iIN = 19; iOUT = 25;
    } else {                                            // insertion: datanorm, inputnet, conv1..4, output
        iDN = 2; iIN = 3; iC1 = 9; iC2 = 13; iC3 = 17; iC4 = 21; iOUT = 25;
    }
    auto FPx = [&](int i) { return (const float*)d_in[i]; };
    const float* dn  = FPx(iDN);
    const float *Wi1 = FPx(iIN+0), *bi1 = FPx(iIN+1), *Wi2 = FPx(iIN+2),
                *bi2 = FPx(iIN+3), *Wi3 = FPx(iIN+4), *bi3 = FPx(iIN+5);
    const float *c1W1 = FPx(iC1+0), *c1b1 = FPx(iC1+1), *c1W2 = FPx(iC1+2), *c1b2 = FPx(iC1+3);
    const float *c2W1 = FPx(iC2+0), *c2b1 = FPx(iC2+1), *c2W2 = FPx(iC2+2), *c2b2 = FPx(iC2+3);
    const float *c3W1 = FPx(iC3+0), *c3b1 = FPx(iC3+1), *c3W2 = FPx(iC3+2), *c3b2 = FPx(iC3+3);
    const float *c4W1 = FPx(iC4+0), *c4b1 = FPx(iC4+1), *c4W2 = FPx(iC4+2), *c4b2 = FPx(iC4+3);
    const float *Wo1 = FPx(iOUT+0), *bo1 = FPx(iOUT+1), *Wo2 = FPx(iOUT+2),
                *bo2 = FPx(iOUT+3), *Wo3 = FPx(iOUT+4), *bo3 = FPx(iOUT+5);

    char* ws = (char*)d_ws;
    int*      meta   = (int*)(ws + OFF_META);
    int*      gsA    = (int*)(ws + OFF_GSA);
    int*      gsB    = (int*)(ws + OFF_GSB);
    int*      batchA = (int*)(ws + OFF_BATCHA);
    int*      batchB = (int*)(ws + OFF_BATCHB);
    int*      deg    = (int*)(ws + OFF_DEG);
    int*      rowptr = (int*)(ws + OFF_ROWPTR);
    int*      match  = (int*)(ws + OFF_MATCH);
    int*      cid    = (int*)(ws + OFF_CID);
    int*      mem1   = (int*)(ws + OFF_MEM1);
    int*      mem2   = (int*)(ws + OFF_MEM2);
    float*    gbuf   = (float*)(ws + OFF_GBUF);
    float*    xn     = (float*)(ws + OFF_XN);
    int*      erow   = (int*)(ws + OFF_EROW);
    int*      ecol   = (int*)(ws + OFF_ECOL);
    float*    ew     = (float*)(ws + OFF_EW);
    unsigned* adj    = (unsigned*)(ws + OFF_ADJ);
    float*    F0     = (float*)(ws + OFF_F0);
    float*    F1     = (float*)(ws + OFF_F1);
    float*    F2     = (float*)(ws + OFF_F2);
    float*    out    = (float*)d_out;

    // init (fresh every call: no cross-call state)
    k_init<<<1, 32, 0, stream>>>(meta, gsA);
    k_copyint<<<N0 / 256, 256, 0, stream>>>(batchA, batch_in, N0);
    hipMemsetAsync(xn, 0, (size_t)N0 * 32 * 4, stream);

    // prepack all WMMA-consumed weights to f16 B-operand layout
    size_t wp = OFF_WP;
    auto packW = [&](const float* W, int K, int Kpad, int N) -> const half_t* {
        half_t* P = (half_t*)(ws + wp);
        int total = Kpad * N;
        k_pack<<<(total + 255) / 256, 256, 0, stream>>>(W, K, Kpad, N, P);
        wp += (size_t)total * 2;
        return P;
    };
    const half_t* Pi1  = packW(Wi1,    5,  32, 128);
    const half_t* Pi2  = packW(Wi2,  128, 128, 128);
    const half_t* Pi3  = packW(Wi3,  128, 128,  64);
    const half_t* Pc1a = packW(c1W1, 128, 128,  96);
    const half_t* Pc1b = packW(c1W2,  96,  96,  64);
    const half_t* Pc2a = packW(c2W1, 256, 256,  96);
    const half_t* Pc2b = packW(c2W2,  96,  96,  64);
    const half_t* Pc3a = packW(c3W1, 384, 384,  96);
    const half_t* Pc3b = packW(c3W2,  96,  96,  64);
    const half_t* Pc4a = packW(c4W1, 512, 512,  96);
    const half_t* Pc4b = packW(c4W2,  96,  96,  64);

    // input MLP: 5 -> 128 -> 128 -> 64 (all lrelu), h0 in F0
    k_scale<<<(N0 * 5 + 255) / 256, 256, 0, stream>>>(xn, x, dn, N0 * 5);
    k_gemm<<<dim3(N0 / 16, 8), 32, 0, stream>>>(xn, 32,     Pi1, bi1, F1, FPITCH, meta,  32, 128, 1);
    k_gemm<<<dim3(N0 / 16, 8), 32, 0, stream>>>(F1, FPITCH, Pi2, bi2, F2, FPITCH, meta, 128, 128, 1);
    k_gemm<<<dim3(N0 / 16, 4), 32, 0, stream>>>(F2, FPITCH, Pi3, bi3, F0, FPITCH, meta, 128,  64, 1);

#define GRAPH_STAGE(Hsrc, Fdim, bCur, gsCur, W1p, b1p, W2p, b2p, Hdst, RES)                          \
    do {                                                                                             \
        hipMemsetAsync(adj, 0, (size_t)N0 * ADJ_W * 4, stream);                                      \
        k_knn<<<N0, 32, 0, stream>>>(Hsrc, FPITCH, Fdim, bCur, gsCur, meta, adj);                    \
        k_deg<<<N0 / 256, 256, 0, stream>>>(adj, deg, meta);                                         \
        k_scan<<<1, 1, 0, stream>>>(deg, rowptr, meta);                                              \
        k_csr<<<N0 / 256, 256, 0, stream>>>(adj, rowptr, erow, ecol, meta);                          \
        k_edgeconv<<<N0 / 16, 32, 0, stream>>>(Hsrc, FPITCH, Fdim, W1p, b1p, W2p, b2p,               \
                                               Hdst, FPITCH, rowptr, ecol, meta);                    \
        if (RES) k_rescopy<<<(N0 * Fdim + 255) / 256, 256, 0, stream>>>(Hdst, FPITCH, Hsrc,          \
                                                                        FPITCH, Fdim, meta);         \
    } while (0)

#define POOL_STAGE(Hn, Ddim, bCur, bNew, gsNew, Hout)                                                \
    do {                                                                                             \
        k_weight<<<MAXE / 256, 256, 0, stream>>>(Hn, FPITCH, Ddim, erow, ecol, deg, ew, meta);       \
        k_graclus<<<1, 1, 0, stream>>>(rowptr, ecol, ew, meta, match, cid, mem1, mem2,               \
                                       bCur, bNew, gsNew);                                           \
        k_pool<<<(N0 * Ddim + 255) / 256, 256, 0, stream>>>(Hn, FPITCH, Ddim, mem1, mem2,            \
                                                            Hout, FPITCH, meta);                     \
    } while (0)

    // stage 1: F=64 -> h_next dim 128, pool -> F0, batchA->batchB
    GRAPH_STAGE(F0,  64, batchA, gsA, Pc1a, c1b1, Pc1b, c1b2, F1, 1);
    POOL_STAGE (F1, 128, batchA, batchB, gsB, F0);

    // stage 2: F=128 -> h_next dim 192, pool -> F0, batchB->batchA
    GRAPH_STAGE(F0, 128, batchB, gsB, Pc2a, c2b1, Pc2b, c2b2, F1, 1);
    POOL_STAGE (F1, 192, batchB, batchA, gsA, F0);

    // stage 3: conv3 (F=192 -> dim 256 in F1), knn again, conv4 (F=256 -> dim 64 in F2)
    GRAPH_STAGE(F0, 192, batchA, gsA, Pc3a, c3b1, Pc3b, c3b2, F1, 1);
    GRAPH_STAGE(F1, 256, batchA, gsA, Pc4a, c4b1, Pc4b, c4b2, F2, 0);
    POOL_STAGE (F2,  64, batchA, batchB, gsB, F0);

    // per-graph max then output MLP (64 -> 64 -> 32 -> 1)
    k_graphmax<<<NG, 64, 0, stream>>>(F0, FPITCH, gsB, gbuf);
    k_outmlp<<<NG, 64, 0, stream>>>(gbuf, Wo1, bo1, Wo2, bo2, Wo3, bo3, out);

#undef GRAPH_STAGE
#undef POOL_STAGE
    (void)out_size; (void)n_in;
}
